// GMMProjector_38826504355893
// MI455X (gfx1250) — compile-verified, matrix operand order checked by reading, not run on hardware
//
#include <hip/hip_runtime.h>

typedef _Float16 h16;
typedef __attribute__((ext_vector_type(2)))  _Float16 h2;
typedef __attribute__((ext_vector_type(16))) _Float16 v16h;
typedef __attribute__((ext_vector_type(8)))  float     v8f;
typedef __attribute__((ext_vector_type(4)))  unsigned int u32x4;

#define SIDE   256
#define NCEN   2048
#define TILE   128     // 128x128 output tile per workgroup
#define KCHUNK 32      // K step = WMMA f16 K
#define ROWP   40      // padded halfs per LDS tile row (80 B, 16B-aligned, bank-spread)

__global__ __launch_bounds__(256) void gmm_proj_wmma(
    const float* __restrict__ centers,   // (2048,3)
    const float* __restrict__ sigmas,    // (2048,)
    const float* __restrict__ amps,      // (2048,)
    const float* __restrict__ rots,      // (B,3,3)
    const float* __restrict__ line,      // (256,)
    float* __restrict__ out)             // (B,256,256) = out[b][y][x]
{
    __shared__ float s_rcx[NCEN];
    __shared__ float s_rcy[NCEN];
    __shared__ float s_inv[NCEN];
    __shared__ float s_amp[NCEN];
    __shared__ __align__(16) h16 s_px[TILE * ROWP];  // B operand (N = x), amp folded in
    __shared__ __align__(16) h16 s_py[TILE * ROWP];  // A operand (M = y)

    const int t     = threadIdx.x;
    const int b     = blockIdx.z;
    const int xbase = blockIdx.x * TILE;
    const int ybase = blockIdx.y * TILE;

    // Rotation rows 0 and 1 (row 2 never used by the projection)
    const float* R = rots + b * 9;
    const float r00 = R[0], r01 = R[1], r02 = R[2];
    const float r10 = R[3], r11 = R[4], r12 = R[5];

    // Per-center precompute into LDS (done once per workgroup)
    for (int i = t; i < NCEN; i += 256) {
        float cx = centers[3*i+0], cy = centers[3*i+1], cz = centers[3*i+2];
        s_rcx[i] = r00*cx + r01*cy + r02*cz;
        s_rcy[i] = r10*cx + r11*cy + r12*cz;
        float s  = sigmas[i];
        s_inv[i] = 0.5f / (s * s);
        s_amp[i] = amps[i];
    }

    // Tile-fill mapping: 2 threads per tile row, 16 centers each
    const int row = t >> 1;          // 0..127
    const int g   = t & 1;           // which 16-column half
    const float lx = line[xbase + row];
    const float ly = line[ybase + row];

    // Wave/lane mapping for WMMA: wave owns 32(y) x 64(x)
    const int wave = t >> 5;
    const int lane = t & 31;
    const int lo   = lane & 15;
    const int hi   = lane >> 4;
    const int yoff = (wave & 3) * 32;   // 2 M-tiles of 16
    const int xoff = (wave >> 2) * 64;  // 4 N-tiles of 16

    v8f zero = {};
    v8f acc[2][4];
    #pragma unroll
    for (int m = 0; m < 2; ++m)
        #pragma unroll
        for (int n = 0; n < 4; ++n) acc[m][n] = zero;

    __syncthreads();

    for (int kk = 0; kk < NCEN; kk += KCHUNK) {
        // ---- fill 128x32 f16 tiles (Px with amplitude folded, Py plain) ----
        const int n0 = kk + g * 16;
        #pragma unroll
        for (int j = 0; j < 16; j += 2) {
            float i0 = s_inv[n0+j], i1 = s_inv[n0+j+1];
            float d0 = lx - s_rcx[n0+j];
            float d1 = lx - s_rcx[n0+j+1];
            float e0 = __expf(-d0*d0*i0) * s_amp[n0+j];
            float e1 = __expf(-d1*d1*i1) * s_amp[n0+j+1];
            *(h2*)&s_px[row*ROWP + g*16 + j] = (h2){(_Float16)e0, (_Float16)e1};
            float f0 = ly - s_rcy[n0+j];
            float f1 = ly - s_rcy[n0+j+1];
            float q0 = __expf(-f0*f0*i0);
            float q1 = __expf(-f1*f1*i1);
            *(h2*)&s_py[row*ROWP + g*16 + j] = (h2){(_Float16)q0, (_Float16)q1};
        }
        __syncthreads();

        // ---- load fragments (two ds_load_b128 each) and run WMMAs ----
        union { v16h v; u32x4 q[2]; } afr[2], bfr[4];
        #pragma unroll
        for (int m = 0; m < 2; ++m) {
            int r = yoff + m*16 + lo;
            afr[m].q[0] = *(const u32x4*)&s_py[r*ROWP + hi*8];       // K =  8*hi + 0..7
            afr[m].q[1] = *(const u32x4*)&s_py[r*ROWP + 16 + hi*8];  // K = 16 + 8*hi + 0..7
        }
        #pragma unroll
        for (int n = 0; n < 4; ++n) {
            int r = xoff + n*16 + lo;
            bfr[n].q[0] = *(const u32x4*)&s_px[r*ROWP + hi*8];
            bfr[n].q[1] = *(const u32x4*)&s_px[r*ROWP + 16 + hi*8];
        }
        #pragma unroll
        for (int m = 0; m < 2; ++m)
            #pragma unroll
            for (int n = 0; n < 4; ++n)
                acc[m][n] = __builtin_amdgcn_wmma_f32_16x16x32_f16(
                    false, afr[m].v, false, bfr[n].v,
                    (short)0, acc[m][n], false, false);
        __syncthreads();
    }

    // ---- write 128x128 tile; lanes 0..15 cover consecutive x -> coalesced ----
    #pragma unroll
    for (int m = 0; m < 2; ++m) {
        #pragma unroll
        for (int n = 0; n < 4; ++n) {
            #pragma unroll
            for (int r = 0; r < 8; ++r) {
                int y = ybase + yoff + m*16 + hi*8 + r;   // D layout: M = r + 8*hi
                int x = xbase + xoff + n*16 + lo;         // D layout: N = lane%16
                out[((size_t)b * SIDE + y) * SIDE + x] = acc[m][n][r];
            }
        }
    }
}

extern "C" void kernel_launch(void* const* d_in, const int* in_sizes, int n_in,
                              void* d_out, int out_size, void* d_ws, size_t ws_size,
                              hipStream_t stream) {
    const float* centers = (const float*)d_in[0];
    const float* sigmas  = (const float*)d_in[1];
    const float* amps    = (const float*)d_in[2];
    const float* rots    = (const float*)d_in[3];
    const float* line    = (const float*)d_in[4];
    float* out = (float*)d_out;

    const int B = in_sizes[3] / 9;   // 64
    dim3 grid(SIDE / TILE, SIDE / TILE, B);
    gmm_proj_wmma<<<grid, 256, 0, stream>>>(centers, sigmas, amps, rots, line, out);
}
